// Hyper_Dec_49108656062673
// MI455X (gfx1250) — matrix operand order repaired and checked by reading, not compiled, not run on hardware
//
#include <hip/hip_runtime.h>
#include <hip/hip_bf16.h>
#include <stdint.h>

typedef __attribute__((ext_vector_type(16))) _Float16 v16h;
typedef __attribute__((ext_vector_type(8)))  _Float16 v8h;
typedef __attribute__((ext_vector_type(8)))  float    v8f;
typedef _Float16 half_t;

struct TapList { int n; int dy[25]; int dx[25]; };

// ---------------------------------------------------------------------------
// WMMA implicit-GEMM convolution.  One wave computes TWO 16(oc) x 16(x) tiles
// (oc tiles 2*blockIdx.y and 2*blockIdx.y+1) sharing one B fragment per K-step.
// in:  NHWC f16 (N,H,W,IC).  out: NHWC f16 (N,Ho,Wo,OC).
// wpk: packed A fragments: [kt][ot][lane][16 halves], K = taps.n*IC (mult of 32)
// Requires IC % 32 == 0 so a 32-wide K-step stays inside one tap.
// lhsdil==2 => stride-2 transposed conv.  act: 0 none, 1 relu, 2 leaky(0.01).
// reluIn: relu applied to loaded input.  res: optional residual added.
// ---------------------------------------------------------------------------
__global__ __launch_bounds__(32) void k_conv_wmma(
    const half_t* __restrict__ in, half_t* __restrict__ out,
    const half_t* __restrict__ wpk, const float* __restrict__ bias,
    const half_t* __restrict__ res,
    int N, int H, int W, int IC, int OC, int Ho, int Wo,
    int pady, int padx, int lhsdil, int act, int reluIn, TapList taps)
{
  const int lane = threadIdx.x & 31;
  const int lid  = lane & 15;
  const int hf   = lane >> 4;
  const int tilesX = Wo >> 4;
  int ts  = blockIdx.x;
  int tx  = ts % tilesX;
  int rem = ts / tilesX;
  int oy  = rem % Ho;
  int n   = rem / Ho;
  int ot0 = blockIdx.y * 2;
  int ox  = (tx << 4) + lid;

  int K     = taps.n * IC;
  int numKT = K >> 5;
  int numOT = OC >> 4;

  v8f acc0 = {};
  v8f acc1 = {};
  int t = 0;          // uniform tap index (no division: IC % 32 == 0)
  int icb = 0;        // k0 base within current tap's IC block
  for (int kt = 0; kt < numKT; ++kt) {
    // A fragments: one aligned 32B load per lane per oc-tile (pre-packed order)
    const half_t* abase = wpk + ((size_t)((size_t)kt * numOT + ot0) * 32 + lane) * 16;
    v16h afrag0 = *(const v16h*)(abase);
    v16h afrag1 = *(const v16h*)(abase + 32 * 16);
    if (kt + 1 < numKT) { // next K-step's A tiles are sequential: prefetch
      __builtin_prefetch(wpk + ((size_t)((size_t)(kt + 1) * numOT + ot0) * 32 + lane) * 16, 0, 3);
    }
    // B fragment: 16 consecutive ic at one spatial tap (shared by both tiles)
    int ic0 = icb + (hf << 4);
    int py  = oy + taps.dy[t] - pady;
    int px  = ox + taps.dx[t] - padx;
    bool valid; int iy, ix;
    if (lhsdil == 2) {
      valid = (py >= 0) && (px >= 0) && !(py & 1) && !(px & 1);
      iy = py >> 1; ix = px >> 1;
      valid = valid && (iy < H) && (ix < W);
    } else {
      valid = (py >= 0) && (py < H) && (px >= 0) && (px < W);
      iy = py; ix = px;
    }
    v16h bfrag = {};
    if (valid) {
      bfrag = *(const v16h*)(in + (((size_t)n * H + iy) * W + ix) * IC + ic0);
      if (reluIn) {
        #pragma unroll
        for (int j = 0; j < 16; ++j)
          bfrag[j] = bfrag[j] > (half_t)0 ? bfrag[j] : (half_t)0;
      }
    }
    acc0 = __builtin_amdgcn_wmma_f32_16x16x32_f16(false, afrag0, false, bfrag,
                                                  (short)0, acc0, false, false);
    acc1 = __builtin_amdgcn_wmma_f32_16x16x32_f16(false, afrag1, false, bfrag,
                                                  (short)0, acc1, false, false);
    icb += 32;
    if (icb == IC) { icb = 0; ++t; }
  }
  size_t pbase = (((size_t)n * Ho + oy) * Wo + ox) * OC;
  #pragma unroll
  for (int r = 0; r < 8; ++r) {
    int m = r + (hf << 3);
    #pragma unroll
    for (int q = 0; q < 2; ++q) {
      int oc = ((ot0 + q) << 4) + m;
      float v = (q ? acc1[r] : acc0[r]) + bias[oc];
      if (act == 1) v = fmaxf(v, 0.f);
      else if (act == 2) v = v > 0.f ? v : 0.01f * v;
      if (res) v += (float)res[pbase + oc];
      out[pbase + oc] = (half_t)v;
    }
  }
}

// ---------------------------------------------------------------------------
// Generic WMMA GEMM:  C[M,N] = A[M,K] * Bt[N,K]^T  (both f16 row-major).
// One wave computes TWO 16x16 tiles: n tiles 2*blockIdx.y, 2*blockIdx.y+1,
// sharing the A fragment.  N % 32 == 0 required.
// Outputs: optional f32 C, optional f16 C (with act), optional f16 transposed C.
// grid = (M/16, N/32, batch); strides sA,sB,sC in elements (sB==0 broadcasts B).
// ---------------------------------------------------------------------------
__global__ __launch_bounds__(32) void k_gemm_nt(
    const half_t* __restrict__ A, const half_t* __restrict__ Bt,
    float* __restrict__ C32, half_t* __restrict__ C16, half_t* __restrict__ CT16,
    int M, int Nn, int K, int act, long sA, long sB, long sC)
{
  const int lane = threadIdx.x & 31;
  const int lid  = lane & 15;
  const int hf   = lane >> 4;
  int mt = blockIdx.x, nt0 = blockIdx.y * 2, b = blockIdx.z;
  const half_t* Ab = A  + (size_t)b * sA;
  const half_t* Bb = Bt + (size_t)b * sB;
  int row  = (mt << 4) + lid;
  int col0 = (nt0 << 4) + lid;
  v8f acc0 = {};
  v8f acc1 = {};
  for (int kt = 0; kt < (K >> 5); ++kt) {
    int ka = (kt << 5) + (hf << 4);
    v8h a0 = *(const v8h*)(Ab + (size_t)row * K + ka);
    v8h a1 = *(const v8h*)(Ab + (size_t)row * K + ka + 8);
    v16h av;
    #pragma unroll
    for (int j = 0; j < 8; ++j) { av[j] = a0[j]; av[j + 8] = a1[j]; }
    int kb = (kt << 5) + (hf << 4);
    v16h bv0 = *(const v16h*)(Bb + (size_t)col0 * K + kb);
    v16h bv1 = *(const v16h*)(Bb + (size_t)(col0 + 16) * K + kb);
    acc0 = __builtin_amdgcn_wmma_f32_16x16x32_f16(false, av, false, bv0,
                                                  (short)0, acc0, false, false);
    acc1 = __builtin_amdgcn_wmma_f32_16x16x32_f16(false, av, false, bv1,
                                                  (short)0, acc1, false, false);
  }
  #pragma unroll
  for (int r = 0; r < 8; ++r) {
    int mr = (mt << 4) + r + (hf << 3);
    #pragma unroll
    for (int q = 0; q < 2; ++q) {
      int nc = ((nt0 + q) << 4) + lid;
      float v = q ? acc1[r] : acc0[r];
      if (act == 1) v = fmaxf(v, 0.f);
      if (C32)  C32 [(size_t)b * sC + (size_t)mr * Nn + nc] = v;
      if (C16)  C16 [(size_t)b * sC + (size_t)mr * Nn + nc] = (half_t)v;
      if (CT16) CT16[(size_t)b * sC + (size_t)nc * M  + mr] = (half_t)v;
    }
  }
}

// ---------------------------------------------------------------------------
// Weight packing: fp32 weights -> f16 A-fragment layout.
// mode 0: OIHW, taps row-major (t = ky*KW+kx), src = (oc*IC+ic)*T + t
// mode 1: transposed 5x5 conv, src = (ic*OC+oc)*25 + (24-t)
// ---------------------------------------------------------------------------
__global__ __launch_bounds__(32) void k_pack_w(
    const float* __restrict__ src, half_t* __restrict__ dst,
    int OC, int IC, int T, int numOT, int ktBase, int mode)
{
  int lane = threadIdx.x & 31;
  int ktL = blockIdx.x, ot = blockIdx.y;
  int m = lane & 15;
  int hfo = (lane >> 4) * 16;
  v16h vv;
  #pragma unroll
  for (int j = 0; j < 16; ++j) {
    int kl = (ktL << 5) + j + hfo;
    int t = kl / IC;
    int ic = kl - t * IC;
    int oc = (ot << 4) + m;
    size_t sidx = (mode == 0) ? ((size_t)(oc * IC + ic) * T + t)
                              : ((size_t)(ic * OC + oc) * 25 + (24 - t));
    vv[j] = (half_t)src[sidx];
  }
  *(v16h*)(dst + ((size_t)((size_t)(ktBase + ktL) * numOT + ot) * 32 + lane) * 16) = vv;
}

__global__ void k_packWt(const float* W, half_t* Wt, int K) { // Wt[n][k]=W[k][n]
  int i = blockIdx.x * 256 + threadIdx.x;
  if (i >= K * K) return;
  int n = i / K, k = i - n * K;
  Wt[(size_t)n * K + k] = (half_t)W[(size_t)k * K + n];
}

__global__ void k_add3f(const float* a, const float* b, const float* c, float* o, int n) {
  int i = blockIdx.x * 256 + threadIdx.x;
  if (i < n) o[i] = a[i] + b[i] + c[i];
}

// ------------------------- elementwise helpers ------------------------------
__global__ void k_add2(const half_t* a, const half_t* b, half_t* o, size_t n) {
  size_t i = (size_t)blockIdx.x * 256 + threadIdx.x;
  if (i < n) o[i] = (half_t)((float)a[i] + (float)b[i]);
}
__global__ void k_addinp(half_t* acc, const half_t* t, size_t n) {
  size_t i = (size_t)blockIdx.x * 256 + threadIdx.x;
  if (i < n) acc[i] = (half_t)((float)acc[i] + (float)t[i]);
}
__global__ void k_add_scaled(half_t* out, const half_t* a, const half_t* h,
                             const float* sc, size_t n, int perN, int C) {
  size_t i = (size_t)blockIdx.x * 256 + threadIdx.x;
  if (i >= n) return;
  int c = (int)(i % C);
  int nb = (int)(i / perN);
  out[i] = (half_t)((float)a[i] + (float)h[i] * sc[nb * C + c]);
}
__global__ void k_combine(half_t* o, const half_t* ca2, const half_t* hy,
                          const half_t* sh, const half_t* gc,
                          const float* beta, const float* gamma, size_t n) {
  size_t i = (size_t)blockIdx.x * 256 + threadIdx.x;
  if (i >= n) return;
  o[i] = (half_t)(beta[0] * ((float)ca2[i] + (float)hy[i]) +
                  gamma[0] * (float)sh[i] + (float)gc[i]);
}
__global__ void k_shuffle(const half_t* sp, half_t* o, int N, int Hs, int Ws, int C) {
  size_t i = (size_t)blockIdx.x * 256 + threadIdx.x;
  size_t tot = (size_t)N * Hs * Ws * 4 * C;
  if (i >= tot) return;
  int c = (int)(i % C); size_t t = i / C;
  int x2 = (int)(t % (2 * Ws)); t /= (2 * Ws);
  int y2 = (int)(t % (2 * Hs)); int n = (int)(t / (2 * Hs));
  int y = y2 >> 1, ii = y2 & 1, x = x2 >> 1, jj = x2 & 1;
  o[i] = sp[(((size_t)n * Hs + y) * Ws + x) * (4 * C) + (c * 4 + ii * 2 + jj)];
}
__global__ void k_nchw2nhwc(const float* x, half_t* o, int N, int C, int H, int W) {
  size_t i = (size_t)blockIdx.x * 256 + threadIdx.x;
  size_t tot = (size_t)N * H * W * C;
  if (i >= tot) return;
  int c = (int)(i % C); size_t t = i / C;
  int xx = (int)(t % W); t /= W;
  int yy = (int)(t % H); int n = (int)(t / H);
  o[i] = (half_t)x[(((size_t)n * C + c) * H + yy) * W + xx];
}
__global__ void k_nhwc2nchw(const half_t* in, float* o, int N, int C, int H, int W) {
  size_t i = (size_t)blockIdx.x * 256 + threadIdx.x;
  size_t tot = (size_t)N * C * H * W;
  if (i >= tot) return;
  int xx = (int)(i % W); size_t t = i / W;
  int yy = (int)(t % H); t /= H;
  int c = (int)(t % C); int n = (int)(t / C);
  o[i] = (float)in[(((size_t)n * H + yy) * W + xx) * C + c];
}

// ----------------------- SE / coord-att small kernels -----------------------
__global__ void k_pool_hw(const half_t* in, float* out, int HW, int C) {
  int n = blockIdx.x, c = threadIdx.x;
  const half_t* p = in + (size_t)n * HW * C + c;
  float s = 0.f;
  for (int i = 0; i < HW; ++i) s += (float)p[(size_t)i * C];
  out[n * C + c] = s / (float)HW;
}
__global__ void k_dense(const float* in, const float* w, const float* b,
                        float* out, int IC, int OC, int mode) {
  int n = blockIdx.x, oc = threadIdx.x;
  float a = b[oc];
  for (int ic = 0; ic < IC; ++ic) a += w[oc * IC + ic] * in[n * IC + ic];
  if (mode == 1) a = fmaxf(a, 0.f);
  else a = 1.f / (1.f + __expf(-a));
  out[n * OC + oc] = a;
}
__global__ void k_ca_pool(const half_t* x, float* y, int H, int W, int C) {
  int n = blockIdx.x, pos = blockIdx.y, c = threadIdx.x;
  float s = 0.f;
  if (pos < H) {
    for (int xx = 0; xx < W; ++xx) s += (float)x[(((size_t)n * H + pos) * W + xx) * C + c];
    s /= (float)W;
  } else {
    int xx = pos - H;
    for (int yy = 0; yy < H; ++yy) s += (float)x[(((size_t)n * H + yy) * W + xx) * C + c];
    s /= (float)H;
  }
  y[((size_t)n * gridDim.y + pos) * C + c] = s;
}
__global__ void k_ca_c1(const float* y, float* y2, const float* w, const float* b,
                        const float* g, const float* bb, int C, int P) {
  int n = blockIdx.x, pos = blockIdx.y, mc = threadIdx.x; // 8 threads
  const float* yp = y + ((size_t)n * P + pos) * C;
  float a = b[mc];
  for (int ic = 0; ic < C; ++ic) a += w[mc * C + ic] * yp[ic];
  a = a * g[mc] + bb[mc];
  float r6 = fminf(fmaxf(a + 3.f, 0.f), 6.f);
  y2[((size_t)n * P + pos) * 8 + mc] = a * r6 / 6.f;
}
__global__ void k_ca_apply(const half_t* x, half_t* o, const float* y2,
                           const float* chw, const float* chb,
                           const float* cww, const float* cwb,
                           int N, int H, int W, int C) {
  size_t i = (size_t)blockIdx.x * 256 + threadIdx.x;
  size_t tot = (size_t)N * H * W * C;
  if (i >= tot) return;
  int c = (int)(i % C); size_t t = i / C;
  int xx = (int)(t % W); t /= W;
  int yy = (int)(t % H); int n = (int)(t / H);
  int P = H + W;
  const float* yh = y2 + ((size_t)n * P + yy) * 8;
  const float* yw = y2 + ((size_t)n * P + H + xx) * 8;
  float ah = chb[c], aw = cwb[c];
  #pragma unroll
  for (int m = 0; m < 8; ++m) { ah += chw[c * 8 + m] * yh[m]; aw += cww[c * 8 + m] * yw[m]; }
  ah = 1.f / (1.f + __expf(-ah));
  aw = 1.f / (1.f + __expf(-aw));
  o[i] = (half_t)((float)x[i] * ah * aw);
}

// ------------------------------ GAT kernels ---------------------------------
__global__ void k_rowmask(const float* adj, unsigned* mask, int Ncols) {
  int wid = threadIdx.x >> 5, lane = threadIdx.x & 31;
  int row = blockIdx.x * (blockDim.x >> 5) + wid;
  int b = blockIdx.y;
  const float* ar = adj + ((size_t)b * Ncols + row) * Ncols;
  float s = 0.f;
  for (int j = lane; j < Ncols; j += 32) s += ar[j];
  for (int m = 16; m; m >>= 1) s += __shfl_xor(s, m, 32);
  float den = s + 1e-10f;
  unsigned* mr = mask + ((size_t)b * Ncols + row) * (Ncols / 32);
  for (int w = 0; w < Ncols / 32; ++w) {
    bool pred = (ar[w * 32 + lane] / den) > 0.f;
    unsigned long long bl = __ballot(pred);
    if (lane == 0) mr[w] = (unsigned)bl;
  }
}
__global__ void k_uv(const half_t* Wh, const float* a, float* u, float* v, int M, int C) {
  int wid = threadIdx.x >> 5, lane = threadIdx.x & 31;
  int row = blockIdx.x * (blockDim.x >> 5) + wid;
  int b = blockIdx.y;
  const half_t* wr = Wh + ((size_t)b * M + row) * C;
  float su = 0.f, sv = 0.f;
  for (int c = lane; c < C; c += 32) {
    float w = (float)wr[c];
    su += w * a[c];
    sv += w * a[C + c];
  }
  for (int m = 16; m; m >>= 1) { su += __shfl_xor(su, m, 32); sv += __shfl_xor(sv, m, 32); }
  if (lane == 0) { u[(size_t)b * M + row] = su; v[(size_t)b * M + row] = sv; }
}
__global__ void k_att_softmax(const float* u, const float* v, const unsigned* mask,
                              half_t* att, int Ncols) {
  int wid = threadIdx.x >> 5, lane = threadIdx.x & 31;
  int row = blockIdx.x * (blockDim.x >> 5) + wid;
  int b = blockIdx.y;
  const unsigned* mr = mask + ((size_t)b * Ncols + row) * (Ncols / 32);
  const float* vb = v + (size_t)b * Ncols;
  float ui = u[(size_t)b * Ncols + row];
  float mx = -3.0e38f; int cnt = 0;
  for (int j = lane; j < Ncols; j += 32) {
    if ((mr[j >> 5] >> (j & 31)) & 1u) {
      float z = ui + vb[j];
      float e = z > 0.f ? z : 0.2f * z;
      mx = fmaxf(mx, e); ++cnt;
    }
  }
  for (int m = 16; m; m >>= 1) { mx = fmaxf(mx, __shfl_xor(mx, m, 32)); cnt += __shfl_xor(cnt, m, 32); }
  half_t* ar = att + ((size_t)b * Ncols + row) * Ncols;
  if (cnt == 0) { // all masked -> reference softmax over constant = uniform
    float p = 1.0f / (float)Ncols;
    for (int j = lane; j < Ncols; j += 32) ar[j] = (half_t)p;
    return;
  }
  float sum = 0.f;
  for (int j = lane; j < Ncols; j += 32) {
    float p = 0.f;
    if ((mr[j >> 5] >> (j & 31)) & 1u) {
      float z = ui + vb[j];
      float e = z > 0.f ? z : 0.2f * z;
      p = __expf(e - mx);
    }
    ar[j] = (half_t)p;
    sum += p;
  }
  for (int m = 16; m; m >>= 1) sum += __shfl_xor(sum, m, 32);
  float inv = 1.f / sum;
  for (int j = lane; j < Ncols; j += 32) ar[j] = (half_t)((float)ar[j] * inv);
}

// ===========================================================================
// Host orchestration
// ===========================================================================
enum {
  I_A1_BNB = 0, I_A1_BNG, I_A1_C1B, I_A1_C1W, I_A1_CHB, I_A1_CHW, I_A1_CWB, I_A1_CWW,
  I_A2_BNB, I_A2_BNG, I_A2_C1B, I_A2_C1W, I_A2_CHB, I_A2_CHW, I_A2_CWB, I_A2_CWW,
  I_BETA1,
  I_BA_HWB, I_BA_HWW, I_BA_SQB, I_BA_SQW, I_BA_VWB, I_BA_VWW,
  I_BB_HWB, I_BB_HWW, I_BB_SQB, I_BB_SQW, I_BB_VWB, I_BB_VWW,
  I_CONV0B, I_CONV0W,
  I_GAMMA1,
  I_GW1, I_GW2, I_GA1, I_GA2,
  I_H1_C1B, I_H1_C1W, I_H1_C2B, I_H1_C2W, I_H1_D1B, I_H1_D1W, I_H1_D2B, I_H1_D2W, I_H1_TLB, I_H1_TLW,
  I_H2_C1B, I_H2_C1W, I_H2_C2B, I_H2_C2W, I_H2_D1B, I_H2_D1W, I_H2_D2B, I_H2_D2W, I_H2_TLB, I_H2_TLW,
  I_SPB, I_SPW,
  I_UP1B, I_UP1W, I_UP2B, I_UP2W,
  I_X
};

static TapList mkRect(int KH, int KW) {
  TapList t; t.n = KH * KW;
  for (int i = 0; i < t.n; ++i) { t.dy[i] = i / KW; t.dx[i] = i % KW; }
  return t;
}
static TapList mkAC() { // fused 3x3(pad1,1) + 1x3(pad0,1) + 3x1(pad1,0), global pad (1,1)
  TapList t; t.n = 15; int i = 0;
  for (int ky = 0; ky < 3; ++ky) for (int kx = 0; kx < 3; ++kx) { t.dy[i] = ky; t.dx[i] = kx; ++i; }
  for (int kx = 0; kx < 3; ++kx) { t.dy[i] = 1; t.dx[i] = kx; ++i; }
  for (int ky = 0; ky < 3; ++ky) { t.dy[i] = ky; t.dx[i] = 1; ++i; }
  return t;
}

extern "C" void kernel_launch(void* const* d_in, const int* in_sizes, int n_in,
                              void* d_out, int out_size, void* d_ws, size_t ws_size,
                              hipStream_t stream) {
  (void)in_sizes; (void)n_in; (void)out_size; (void)ws_size;
  auto F = [&](int i) { return (const float*)d_in[i]; };

  const TapList T33 = mkRect(3, 3);
  const TapList T55 = mkRect(5, 5);
  const TapList TAC = mkAC();

  // ---- workspace bump allocator ----
  char* wsp = (char*)d_ws;
  size_t off = 0;
  auto allocB = [&](size_t bytes) -> void* {
    void* p = wsp + off; off = (off + bytes + 255) & ~(size_t)255; return p;
  };
  auto aH = [&](size_t e) { return (half_t*)allocB(e * sizeof(half_t)); };
  auto aF = [&](size_t e) { return (float*) allocB(e * sizeof(float)); };
  auto aU = [&](size_t e) { return (unsigned*)allocB(e * sizeof(unsigned)); };

  const size_t S16 = (size_t)8 * 16 * 16 * 192;
  const size_t S32 = (size_t)8 * 32 * 32 * 192;
  const size_t S64 = (size_t)8 * 64 * 64 * 192;
  const size_t PW33 = (size_t)192 * 1728;   // one 192->192 3x3 packed conv
  const size_t PWAC = (size_t)192 * 2880;   // one fused ac-block conv

  // packed weights
  half_t* pw_conv0 = aH((size_t)192 * 1152);
  half_t* pw_h1c1 = aH(6 * PW33); half_t* pw_h1c2 = aH(6 * PW33); half_t* pw_h1tl = aH(PW33);
  half_t* pw_h2c1 = aH(6 * PW33); half_t* pw_h2c2 = aH(6 * PW33); half_t* pw_h2tl = aH(PW33);
  half_t* pw_acA = aH(3 * PWAC); half_t* pw_acB = aH(3 * PWAC);
  half_t* pw_up1 = aH((size_t)192 * 4800); half_t* pw_up2 = aH((size_t)192 * 4800);
  half_t* pw_sp  = aH((size_t)768 * 1728);
  half_t* w1t = aH(36864); half_t* w2t = aH(36864);
  float* cbA = aF(576); float* cbB = aF(576);

  // activations
  half_t* xin  = aH((size_t)8 * 16 * 16 * 128);
  half_t* x0 = aH(S16); half_t* ca1 = aH(S16); half_t* rb16 = aH(S16);
  half_t* h16a = aH(S16); half_t* h16b = aH(S16); half_t* rgo = aH(S16); half_t* x1 = aH(S16);
  float* ybuf = aF((size_t)8 * 64 * 192); float* ybuf2 = aF((size_t)8 * 64 * 8);
  float* sv1 = aF(8 * 192); float* sv2 = aF(8 * 192);
  half_t* x3 = aH(S32); half_t* t1a = aH(S32); half_t* t1b = aH(S32);
  half_t* high1 = aH(S32);
  half_t* hyr = aH(S32); half_t* hyh1 = aH(S32); half_t* hyh2 = aH(S32); half_t* hyper2 = aH(S32);
  half_t* ca2 = aH(S32); half_t* sp768 = aH((size_t)8 * 256 * 768);
  half_t* shuf = aH(S32); half_t* high1b = aH(S32);
  float* adjraw = aF((size_t)8 * 1024 * 1024);
  unsigned* maskb = aU((size_t)8 * 1024 * 32);
  float* uarr = aF(8192); float* varr = aF(8192);
  half_t* attb = aH((size_t)8 * 1024 * 1024);
  half_t* Wh = aH((size_t)8 * 1024 * 192); half_t* WhT = aH((size_t)8 * 192 * 1024);
  half_t* gc1 = aH((size_t)8 * 1024 * 192); half_t* gc2 = aH((size_t)8 * 1024 * 192);
  half_t* x4 = aH(S64); half_t* t2a = aH(S64); half_t* t2b = aH(S64); half_t* h2acc = aH(S64);

  // ---- helpers ----
  auto pack = [&](const float* src, half_t* dst, int OC, int IC, int T, int ktBase, int mode) {
    dim3 g((unsigned)(T * IC / 32), (unsigned)(OC / 16));
    k_pack_w<<<g, 32, 0, stream>>>(src, dst, OC, IC, T, OC / 16, ktBase, mode);
  };
  auto conv = [&](const half_t* in, half_t* out, const half_t* wpk, const float* bias,
                  const half_t* res, int N, int H, int W, int IC, int OC, int Ho, int Wo,
                  int pady, int padx, int dil, int act, int rin, const TapList& tp) {
    dim3 g((unsigned)(N * Ho * (Wo / 16)), (unsigned)(OC / 32));
    k_conv_wmma<<<g, 32, 0, stream>>>(in, out, wpk, bias, res, N, H, W, IC, OC,
                                      Ho, Wo, pady, padx, dil, act, rin, tp);
  };
  auto gemm = [&](const half_t* A, const half_t* Bt, float* C32, half_t* C16, half_t* CT16,
                  int M, int Nn, int K, int act, long sA, long sB, long sC) {
    dim3 g((unsigned)(M / 16), (unsigned)(Nn / 32), 8);
    k_gemm_nt<<<g, 32, 0, stream>>>(A, Bt, C32, C16, CT16, M, Nn, K, act, sA, sB, sC);
  };
  auto ew = [&](size_t n) { return dim3((unsigned)((n + 255) / 256)); };

  auto coordatt = [&](const half_t* x, half_t* o, int H, int W,
                      const float* c1w, const float* c1b, const float* bng, const float* bnb,
                      const float* chw, const float* chb, const float* cww, const float* cwb) {
    k_ca_pool<<<dim3(8, H + W), 192, 0, stream>>>(x, ybuf, H, W, 192);
    k_ca_c1<<<dim3(8, H + W), 8, 0, stream>>>(ybuf, ybuf2, c1w, c1b, bng, bnb, 192, H + W);
    size_t S = (size_t)8 * H * W * 192;
    k_ca_apply<<<ew(S), 256, 0, stream>>>(x, o, ybuf2, chw, chb, cww, cwb, 8, H, W, 192);
  };

  auto resgroup = [&](const half_t* xin0, half_t* rbuf, half_t* h1, half_t* h2, half_t* outb,
                      const half_t* pwc1, const half_t* pwc2, const half_t* pwtail,
                      const float* c1b, const float* c2b, const float* d1w, const float* d1b,
                      const float* d2w, const float* d2b, const float* tailb, int Hh, int Ww) {
    size_t S = (size_t)8 * Hh * Ww * 192;
    const half_t* r = xin0;
    for (int i = 0; i < 6; ++i) {
      conv(r, h1, pwc1 + (size_t)i * PW33, c1b + i * 192, nullptr,
           8, Hh, Ww, 192, 192, Hh, Ww, 1, 1, 1, 1, 0, T33);
      conv(h1, h2, pwc2 + (size_t)i * PW33, c2b + i * 192, nullptr,
           8, Hh, Ww, 192, 192, Hh, Ww, 1, 1, 1, 0, 0, T33);
      k_pool_hw<<<8, 192, 0, stream>>>(h2, sv1, Hh * Ww, 192);
      k_dense<<<8, 192, 0, stream>>>(sv1, d1w + (size_t)i * 36864, d1b + i * 192, sv2, 192, 192, 1);
      k_dense<<<8, 192, 0, stream>>>(sv2, d2w + (size_t)i * 36864, d2b + i * 192, sv1, 192, 192, 2);
      k_add_scaled<<<ew(S), 256, 0, stream>>>(rbuf, r, h2, sv1, S, Hh * Ww * 192, 192);
      r = rbuf;
    }
    conv(rbuf, outb, pwtail, tailb, xin0, 8, Hh, Ww, 192, 192, Hh, Ww, 1, 1, 1, 0, 0, T33);
  };

  // ================= phase 0: pack weights ================================
  pack(F(I_CONV0W), pw_conv0, 192, 128, 9, 0, 0);
  for (int i = 0; i < 6; ++i) {
    pack(F(I_H1_C1W) + (size_t)i * 331776, pw_h1c1 + (size_t)i * PW33, 192, 192, 9, 0, 0);
    pack(F(I_H1_C2W) + (size_t)i * 331776, pw_h1c2 + (size_t)i * PW33, 192, 192, 9, 0, 0);
    pack(F(I_H2_C1W) + (size_t)i * 331776, pw_h2c1 + (size_t)i * PW33, 192, 192, 9, 0, 0);
    pack(F(I_H2_C2W) + (size_t)i * 331776, pw_h2c2 + (size_t)i * PW33, 192, 192, 9, 0, 0);
  }
  pack(F(I_H1_TLW), pw_h1tl, 192, 192, 9, 0, 0);
  pack(F(I_H2_TLW), pw_h2tl, 192, 192, 9, 0, 0);
  for (int i = 0; i < 3; ++i) { // fused ac blocks: sq(kt 0..53) hw(54..71) vw(72..89)
    pack(F(I_BA_SQW) + (size_t)i * 331776, pw_acA + (size_t)i * PWAC, 192, 192, 9, 0, 0);
    pack(F(I_BA_HWW) + (size_t)i * 110592, pw_acA + (size_t)i * PWAC, 192, 192, 3, 54, 0);
    pack(F(I_BA_VWW) + (size_t)i * 110592, pw_acA + (size_t)i * PWAC, 192, 192, 3, 72, 0);
    pack(F(I_BB_SQW) + (size_t)i * 331776, pw_acB + (size_t)i * PWAC, 192, 192, 9, 0, 0);
    pack(F(I_BB_HWW) + (size_t)i * 110592, pw_acB + (size_t)i * PWAC, 192, 192, 3, 54, 0);
    pack(F(I_BB_VWW) + (size_t)i * 110592, pw_acB + (size_t)i * PWAC, 192, 192, 3, 72, 0);
  }
  pack(F(I_UP1W), pw_up1, 192, 192, 25, 0, 1);
  pack(F(I_UP2W), pw_up2, 192, 192, 25, 0, 1);
  pack(F(I_SPW), pw_sp, 768, 192, 9, 0, 0);
  k_packWt<<<144, 256, 0, stream>>>(F(I_GW1), w1t, 192);
  k_packWt<<<144, 256, 0, stream>>>(F(I_GW2), w2t, 192);
  k_add3f<<<3, 256, 0, stream>>>(F(I_BA_SQB), F(I_BA_HWB), F(I_BA_VWB), cbA, 576);
  k_add3f<<<3, 256, 0, stream>>>(F(I_BB_SQB), F(I_BB_HWB), F(I_BB_VWB), cbB, 576);

  // ================= phase 1: forward =====================================
  k_nchw2nhwc<<<ew((size_t)8 * 16 * 16 * 128), 256, 0, stream>>>(F(I_X), xin, 8, 128, 16, 16);
  conv(xin, x0, pw_conv0, F(I_CONV0B), nullptr, 8, 16, 16, 128, 192, 16, 16, 1, 1, 1, 0, 0, T33);

  coordatt(x0, ca1, 16, 16, F(I_A1_C1W), F(I_A1_C1B), F(I_A1_BNG), F(I_A1_BNB),
           F(I_A1_CHW), F(I_A1_CHB), F(I_A1_CWW), F(I_A1_CWB));
  resgroup(x0, rb16, h16a, h16b, rgo, pw_h1c1, pw_h1c2, pw_h1tl,
           F(I_H1_C1B), F(I_H1_C2B), F(I_H1_D1W), F(I_H1_D1B),
           F(I_H1_D2W), F(I_H1_D2B), F(I_H1_TLB), 16, 16);
  k_add2<<<ew(S16), 256, 0, stream>>>(ca1, rgo, x1, S16);

  conv(x1, x3, pw_up1, F(I_UP1B), nullptr, 8, 16, 16, 192, 192, 32, 32, 2, 2, 2, 2, 0, T55);

  conv(x3, t1a, pw_acA, cbA, nullptr, 8, 32, 32, 192, 192, 32, 32, 1, 1, 1, 0, 1, TAC);
  k_add2<<<ew(S32), 256, 0, stream>>>(x3, t1a, high1, S32);
  conv(t1a, t1b, pw_acA + PWAC, cbA + 192, nullptr, 8, 32, 32, 192, 192, 32, 32, 1, 1, 1, 0, 1, TAC);
  k_addinp<<<ew(S32), 256, 0, stream>>>(high1, t1b, S32);
  conv(t1b, t1a, pw_acA + 2 * PWAC, cbA + 384, nullptr, 8, 32, 32, 192, 192, 32, 32, 1, 1, 1, 0, 1, TAC);
  k_addinp<<<ew(S32), 256, 0, stream>>>(high1, t1a, S32);

  resgroup(high1, hyr, hyh1, hyh2, hyper2, pw_h2c1, pw_h2c2, pw_h2tl,
           F(I_H2_C1B), F(I_H2_C2B), F(I_H2_D1W), F(I_H2_D1B),
           F(I_H2_D2W), F(I_H2_D2B), F(I_H2_TLB), 32, 32);

  // ---- GAT on feats = high1 (NHWC f16 == [b][1024][192]) ----
  {
    const long sF = 1024 * 192, sAdj = 1024 * 1024;
    gemm(high1, high1, adjraw, nullptr, nullptr, 1024, 1024, 192, 0, sF, sF, sAdj);
    k_rowmask<<<dim3(128, 8), 256, 0, stream>>>(adjraw, maskb, 1024);
    // layer 1
    gemm(high1, w1t, nullptr, Wh, WhT, 1024, 192, 192, 0, sF, 0, sF);
    k_uv<<<dim3(128, 8), 256, 0, stream>>>(Wh, F(I_GA1), uarr, varr, 1024, 192);
    k_att_softmax<<<dim3(128, 8), 256, 0, stream>>>(uarr, varr, maskb, attb, 1024);
    gemm(attb, WhT, nullptr, gc1, nullptr, 1024, 192, 1024, 1, sAdj, sF, sF);
    // layer 2  (relu(elu(z)) == relu(z))
    gemm(gc1, w2t, nullptr, Wh, WhT, 1024, 192, 192, 0, sF, 0, sF);
    k_uv<<<dim3(128, 8), 256, 0, stream>>>(Wh, F(I_GA2), uarr, varr, 1024, 192);
    k_att_softmax<<<dim3(128, 8), 256, 0, stream>>>(uarr, varr, maskb, attb, 1024);
    gemm(attb, WhT, nullptr, gc2, nullptr, 1024, 192, 1024, 1, sAdj, sF, sF);
  }

  coordatt(high1, ca2, 32, 32, F(I_A2_C1W), F(I_A2_C1B), F(I_A2_BNG), F(I_A2_BNB),
           F(I_A2_CHW), F(I_A2_CHB), F(I_A2_CWW), F(I_A2_CWB));

  conv(x1, sp768, pw_sp, F(I_SPB), nullptr, 8, 16, 16, 192, 768, 16, 16, 1, 1, 1, 0, 0, T33);
  k_shuffle<<<ew(S32), 256, 0, stream>>>(sp768, shuf, 8, 16, 16, 192);

  k_combine<<<ew(S32), 256, 0, stream>>>(high1b, ca2, hyper2, shuf, gc2,
                                         F(I_BETA1), F(I_GAMMA1), S32);

  conv(high1b, x4, pw_up2, F(I_UP2B), nullptr, 8, 32, 32, 192, 192, 64, 64, 2, 2, 2, 2, 0, T55);

  conv(x4, t2a, pw_acB, cbB, nullptr, 8, 64, 64, 192, 192, 64, 64, 1, 1, 1, 0, 1, TAC);
  k_add2<<<ew(S64), 256, 0, stream>>>(x4, t2a, h2acc, S64);
  conv(t2a, t2b, pw_acB + PWAC, cbB + 192, nullptr, 8, 64, 64, 192, 192, 64, 64, 1, 1, 1, 0, 1, TAC);
  k_addinp<<<ew(S64), 256, 0, stream>>>(h2acc, t2b, S64);
  conv(t2b, t2a, pw_acB + 2 * PWAC, cbB + 384, nullptr, 8, 64, 64, 192, 192, 64, 64, 1, 1, 1, 0, 1, TAC);
  k_addinp<<<ew(S64), 256, 0, stream>>>(h2acc, t2a, S64);

  k_nhwc2nchw<<<ew(S64), 256, 0, stream>>>(h2acc, (float*)d_out, 8, 192, 64, 64);
}
